// multihead_self_attention_51900384804981
// MI455X (gfx1250) — matrix operand order repaired and checked by reading, not compiled, not run on hardware
//
#include <hip/hip_runtime.h>
#include <hip/hip_bf16.h>
#include <math.h>

typedef __bf16 bf16;
typedef __attribute__((ext_vector_type(16))) __bf16 v16bf;
typedef __attribute__((ext_vector_type(8)))  float  v8f;
typedef int v4i_vs __attribute__((vector_size(16)));   // matches async-LDS builtin params

#define B_  4
#define S_  2048
#define D_  1024
#define H_  16
#define DK_ 64
#define QT_ (S_/16)      // 128 query tiles per (b,h)
#define NKSTEP_ (D_/32)  // 32 K-slabs of depth 32

union Frag16 { v16bf v; unsigned u[8]; uint4 q[2]; };

__device__ __forceinline__ unsigned ldg_u32(const bf16* p) {
  return *reinterpret_cast<const unsigned*>(p);
}

// ---- CDNA5 async global->LDS path (guarded; falls back to sync staging) ----
#if defined(__has_builtin)
#if __has_builtin(__builtin_amdgcn_global_load_async_to_lds_b128)
#define HAVE_ASYNC_LDS 1
#endif
#endif

#ifdef HAVE_ASYNC_LDS
__device__ __forceinline__ void async_cp16(const bf16* g, bf16* l) {
  __builtin_amdgcn_global_load_async_to_lds_b128(
      (__attribute__((address_space(1))) v4i_vs*)(g),
      (__attribute__((address_space(3))) v4i_vs*)(l), 0, 0);
}
__device__ __forceinline__ void wait_async_le4() {
#if __has_builtin(__builtin_amdgcn_s_wait_asynccnt)
  __builtin_amdgcn_s_wait_asynccnt(4);
#else
  asm volatile("s_wait_asynccnt 0x4" ::: "memory");
#endif
}
__device__ __forceinline__ void wait_async_le0() {
#if __has_builtin(__builtin_amdgcn_s_wait_asynccnt)
  __builtin_amdgcn_s_wait_asynccnt(0);
#else
  asm volatile("s_wait_asynccnt 0x0" ::: "memory");
#endif
}
#endif

// ---------------------------------------------------------------------------
// fp32 -> bf16 conversion (grid-stride)
// ---------------------------------------------------------------------------
__global__ __launch_bounds__(256) void cvt_f32_bf16(const float* __restrict__ in,
                                                    bf16* __restrict__ out, int n) {
  int i = blockIdx.x * blockDim.x + threadIdx.x;
  int stride = gridDim.x * blockDim.x;
  for (; i < n; i += stride) out[i] = (bf16)in[i];
}

// ---------------------------------------------------------------------------
// Stage one 128x32 bf16 slab of A and of W into LDS.
// 512 16-byte chunks per operand / 256 threads -> 2 chunks each; with async:
// 4 ASYNCcnt increments per wave per stage.
// ---------------------------------------------------------------------------
__device__ __forceinline__ void stage_slab(const bf16* __restrict__ Ag,
                                           const bf16* __restrict__ Wg,
                                           bf16* la, bf16* lb, int tid) {
#pragma unroll
  for (int s = 0; s < 2; ++s) {
    const int c   = tid + s * 256;
    const int row = c >> 2;            // 0..127
    const int ce  = (c & 3) * 8;       // element offset in 32-wide slab
    const bf16* ga = Ag + row * D_ + ce;
    const bf16* gw = Wg + row * D_ + ce;
    bf16* lA = la + row * 32 + ce;
    bf16* lW = lb + row * 32 + ce;
#ifdef HAVE_ASYNC_LDS
    async_cp16(ga, lA);
    async_cp16(gw, lW);
#else
    *(uint4*)lA = *(const uint4*)ga;
    *(uint4*)lW = *(const uint4*)gw;
#endif
  }
}

// ---------------------------------------------------------------------------
// LDS-tiled GEMM:  C[8192,1024] = A[8192,1024] * W[1024,1024]^T
// Workgroup: 128x128 tile, 8 waves in a 4(M) x 2(N) grid, each wave 32x64
// (2x4 of 16x16 WMMA frags). Double-buffered async K-slabs of depth 32.
// MODE 0: bf16 out, V transposed layout  [((b*H+h)*DK + dch)*S + pos]
// MODE 1: RoPE epilogue, bf16 out        [((b*H+h)*S + pos)*DK + dch]
// MODE 2: fp32 out row-major             [r*D + o]
// ---------------------------------------------------------------------------
template<int MODE>
__global__ __launch_bounds__(256) void gemm_bf16_128x128(const bf16* __restrict__ A,
                                                         const bf16* __restrict__ W,
                                                         bf16* __restrict__ outb,
                                                         float* __restrict__ outf) {
  __shared__ __align__(16) bf16 lA[2][128 * 32];
  __shared__ __align__(16) bf16 lB[2][128 * 32];

  const int tid  = threadIdx.x;
  const int lane = tid & 31;
  const int wave = tid >> 5;
  const int half = lane >> 4;
  const int lid  = lane & 15;
  const int wrow = wave & 3;           // 4 waves along M (32 rows each)
  const int wcol = wave >> 2;          // 2 waves along N (64 cols each)

  const int NTB = D_ / 128;            // 8 blocks along N
  const int mt  = blockIdx.x / NTB;
  const int nt  = blockIdx.x % NTB;

  const bf16* Ag = A + (mt * 128) * D_;
  const bf16* Wg = W + (nt * 128) * D_;

  v8f acc[2][4];
#pragma unroll
  for (int mf = 0; mf < 2; ++mf)
#pragma unroll
    for (int nf = 0; nf < 4; ++nf) acc[mf][nf] = {};

  stage_slab(Ag, Wg, lA[0], lB[0], tid);

  for (int it = 0; it < NKSTEP_; ++it) {
    const int cur = it & 1;
#ifdef HAVE_ASYNC_LDS
    if (it + 1 < NKSTEP_) {
      stage_slab(Ag + (it + 1) * 32, Wg + (it + 1) * 32, lA[cur ^ 1], lB[cur ^ 1], tid);
      wait_async_le4();                 // prev slab retired; next stays in flight
    } else {
      wait_async_le0();
    }
#else
    if (it + 1 < NKSTEP_)
      stage_slab(Ag + (it + 1) * 32, Wg + (it + 1) * 32, lA[cur ^ 1], lB[cur ^ 1], tid);
#endif
    __syncthreads();

    // A fragments: rows wrow*32 + mf*16 + lid, 32-deep K
    Frag16 af[2];
#pragma unroll
    for (int mf = 0; mf < 2; ++mf) {
      const int row = wrow * 32 + mf * 16 + lid;
      const bf16* base = &lA[cur][row * 32];
      af[mf].q[0] = *(const uint4*)(base + 8 * half);
      af[mf].q[1] = *(const uint4*)(base + 16 + 8 * half);
    }
#pragma unroll
    for (int nf = 0; nf < 4; ++nf) {
      const int n = wcol * 64 + nf * 16 + lid;
      const bf16* base = &lB[cur][n * 32];
      Frag16 bf_;
      bf_.q[0] = *(const uint4*)(base + 8 * half);
      bf_.q[1] = *(const uint4*)(base + 16 + 8 * half);
#pragma unroll
      for (int mf = 0; mf < 2; ++mf)
        acc[mf][nf] = __builtin_amdgcn_wmma_f32_16x16x32_bf16(
            false, af[mf].v, false, bf_.v, (short)0, acc[mf][nf], false, false);
    }
    __syncthreads();
  }

  // ---- epilogue ----
#pragma unroll
  for (int mf = 0; mf < 2; ++mf) {
#pragma unroll
    for (int nf = 0; nf < 4; ++nf) {
#pragma unroll
      for (int j = 0; j < 8; ++j) {
        const int r = mt * 128 + wrow * 32 + mf * 16 + j + 8 * half;  // b*S + pos
        const int o = nt * 128 + wcol * 64 + nf * 16 + lid;           // out channel
        float val = acc[mf][nf][j];
        if (MODE == 2) {
          outf[r * D_ + o] = val;
        } else {
          const int bb  = r / S_;
          const int pos = r % S_;
          const int hh  = o / DK_;
          const int dch = o % DK_;
          if (MODE == 1) {
            const float partner = __shfl_xor(val, 1);   // pair lives in adjacent lane
            const int   i2   = dch & ~1;
            const float freq = __expf(-9.2103403719761836f * (float)i2 * (1.0f/64.0f));
            const float ang  = (float)pos * freq;
            float sn, cs;
            __sincosf(ang, &sn, &cs);
            const float rot = (dch & 1) ? partner : -partner;
            val = val * cs + rot * sn;
            outb[((bb*H_ + hh)*S_ + pos)*DK_ + dch] = (bf16)val;
          } else {  // MODE 0: transposed V
            outb[((bb*H_ + hh)*DK_ + dch)*S_ + pos] = (bf16)val;
          }
        }
      }
    }
  }
}

// ---------------------------------------------------------------------------
// Flash attention: one wave per (b, h, 16-query tile); 32-key blocks,
// online softmax; scores and P@V both via v_wmma_f32_16x16x32_bf16.
// ---------------------------------------------------------------------------
__global__ __launch_bounds__(256) void attn_kernel(const bf16* __restrict__ Q,
                                                   const bf16* __restrict__ K,
                                                   const bf16* __restrict__ Vt,
                                                   bf16* __restrict__ ctx) {
  __shared__ __align__(16) bf16 plds[8][16 * 32];   // per-wave 16x32 P tile

  const int lane = threadIdx.x & 31;
  const int wave = threadIdx.x >> 5;
  const int wid  = blockIdx.x * 8 + wave;
  const int qt   = wid % QT_;
  const int h    = (wid / QT_) % H_;
  const int bb   = wid / (QT_ * H_);
  const int half = lane >> 4;
  const int lid  = lane & 15;
  const int q0   = qt * 16;

  const int head = (bb * H_ + h) * S_ * DK_;
  const bf16* Qh = Q  + head;
  const bf16* Kh = K  + head;
  const bf16* Vh = Vt + head;                 // [DK][S]

  Frag16 aq0, aq1;
#pragma unroll
  for (int j = 0; j < 8; ++j) {
    const int ka = (j < 4) ? (2*j + 8*half) : (16 + 2*(j-4) + 8*half);
    aq0.u[j] = ldg_u32(Qh + (q0 + lid)*DK_ + ka);
    aq1.u[j] = ldg_u32(Qh + (q0 + lid)*DK_ + 32 + ka);
  }

  v8f o0 = {}, o1 = {}, o2 = {}, o3 = {};
  float rowm[8], rowl[8];
#pragma unroll
  for (int j = 0; j < 8; ++j) { rowm[j] = -3.0e38f; rowl[j] = 0.0f; }

  for (int kb = 0; kb <= q0 + 15; kb += 32) {
    // ---- scores S[16 x 32] ----
    v8f sc[2]; sc[0] = {}; sc[1] = {};
#pragma unroll
    for (int nt2 = 0; nt2 < 2; ++nt2) {
      Frag16 bk0, bk1;
      const int key = kb + nt2*16 + lid;
#pragma unroll
      for (int j = 0; j < 8; ++j) {
        const int kd = (j < 4) ? (2*j + 8*half) : (16 + 2*(j-4) + 8*half);
        bk0.u[j] = ldg_u32(Kh + key*DK_ + kd);
        bk1.u[j] = ldg_u32(Kh + key*DK_ + 32 + kd);
      }
      sc[nt2] = __builtin_amdgcn_wmma_f32_16x16x32_bf16(false, aq0.v, false, bk0.v,
                                                        (short)0, sc[nt2], false, false);
      sc[nt2] = __builtin_amdgcn_wmma_f32_16x16x32_bf16(false, aq1.v, false, bk1.v,
                                                        (short)0, sc[nt2], false, false);
    }

    // ---- scale + causal mask ----
#pragma unroll
    for (int nt2 = 0; nt2 < 2; ++nt2) {
      const int key = kb + nt2*16 + lid;
#pragma unroll
      for (int j = 0; j < 8; ++j) {
        const int q = q0 + j + 8*half;
        const float s = sc[nt2][j] * 0.125f;
        sc[nt2][j] = (key > q) ? -3.0e38f : s;
      }
    }

    // ---- online softmax ----
    float fscale[8];
#pragma unroll
    for (int j = 0; j < 8; ++j) {
      float mx = fmaxf(sc[0][j], sc[1][j]);
#pragma unroll
      for (int off = 1; off < 16; off <<= 1) mx = fmaxf(mx, __shfl_xor(mx, off));
      const float mnew = fmaxf(rowm[j], mx);
      const float f  = __expf(rowm[j] - mnew);
      const float p0 = __expf(sc[0][j] - mnew);
      const float p1 = __expf(sc[1][j] - mnew);
      sc[0][j] = p0; sc[1][j] = p1;
      float ps = p0 + p1;
#pragma unroll
      for (int off = 1; off < 16; off <<= 1) ps += __shfl_xor(ps, off);
      rowl[j] = rowl[j] * f + ps;
      rowm[j] = mnew;
      fscale[j] = f;
    }
#pragma unroll
    for (int j = 0; j < 8; ++j) {
      o0[j] *= fscale[j]; o1[j] *= fscale[j];
      o2[j] *= fscale[j]; o3[j] *= fscale[j];
    }

    // ---- P (C layout) -> A fragment via per-wave LDS tile (in-order LDS) ----
#pragma unroll
    for (int nt2 = 0; nt2 < 2; ++nt2)
#pragma unroll
      for (int j = 0; j < 8; ++j)
        plds[wave][(j + 8*half)*32 + nt2*16 + lid] = (bf16)sc[nt2][j];

    Frag16 pa;
#pragma unroll
    for (int j = 0; j < 8; ++j) {
      const int ka = (j < 4) ? (2*j + 8*half) : (16 + 2*(j-4) + 8*half);
      pa.u[j] = *reinterpret_cast<const unsigned*>(&plds[wave][lid*32 + ka]);
    }

    // ---- O += P @ V ----
#pragma unroll
    for (int c4 = 0; c4 < 4; ++c4) {
      Frag16 bv;
      const int dch = c4*16 + lid;
#pragma unroll
      for (int j = 0; j < 8; ++j) {
        const int kk2 = kb + ((j < 4) ? (2*j + 8*half) : (16 + 2*(j-4) + 8*half));
        bv.u[j] = ldg_u32(Vh + dch*S_ + kk2);
      }
      v8f* op = (c4 == 0) ? &o0 : (c4 == 1) ? &o1 : (c4 == 2) ? &o2 : &o3;
      *op = __builtin_amdgcn_wmma_f32_16x16x32_bf16(false, pa.v, false, bv.v,
                                                    (short)0, *op, false, false);
    }
  }

  // ---- normalize + store context [b, s, h, dk] ----
#pragma unroll
  for (int j = 0; j < 8; ++j) {
    const float inv = 1.0f / rowl[j];
    const int q = q0 + j + 8*half;
    const int base = ((bb*S_ + q)*H_ + h)*DK_;
    ctx[base +  0 + lid] = (bf16)(o0[j] * inv);
    ctx[base + 16 + lid] = (bf16)(o1[j] * inv);
    ctx[base + 32 + lid] = (bf16)(o2[j] * inv);
    ctx[base + 48 + lid] = (bf16)(o3[j] * inv);
  }
}

// ---------------------------------------------------------------------------
extern "C" void kernel_launch(void* const* d_in, const int* in_sizes, int n_in,
                              void* d_out, int out_size, void* d_ws, size_t ws_size,
                              hipStream_t stream) {
  (void)in_sizes; (void)n_in; (void)out_size; (void)ws_size;
  const float* x  = (const float*)d_in[0];
  const float* wq = (const float*)d_in[1];
  const float* wk = (const float*)d_in[2];
  const float* wv = (const float*)d_in[3];
  const float* wo = (const float*)d_in[4];
  float* out = (float*)d_out;

  const int NX = B_ * S_ * D_;          // 8,388,608
  const int NW = D_ * D_;               // 1,048,576

  char* ws = (char*)d_ws;
  bf16* xb  = (bf16*)(ws);                                  // 16 MB
  bf16* wqb = (bf16*)(ws + 16777216);                       //  2 MB
  bf16* wkb = (bf16*)(ws + 18874368);
  bf16* wvb = (bf16*)(ws + 20971520);
  bf16* wob = (bf16*)(ws + 23068672);
  bf16* Qb  = (bf16*)(ws + 25165824);                       // 16 MB
  bf16* Kb  = (bf16*)(ws + 41943040);                       // 16 MB
  bf16* Vt  = (bf16*)(ws + 58720256);                       // 16 MB
  bf16* ctx = (bf16*)(ws + 75497472);                       // 16 MB

  cvt_f32_bf16<<<4096, 256, 0, stream>>>(x,  xb,  NX);
  cvt_f32_bf16<<<1024, 256, 0, stream>>>(wq, wqb, NW);
  cvt_f32_bf16<<<1024, 256, 0, stream>>>(wk, wkb, NW);
  cvt_f32_bf16<<<1024, 256, 0, stream>>>(wv, wvb, NW);
  cvt_f32_bf16<<<1024, 256, 0, stream>>>(wo, wob, NW);

  const int gemmBlocks = (B_*S_/128) * (D_/128);            // 64*8 = 512
  gemm_bf16_128x128<1><<<gemmBlocks, 256, 0, stream>>>(xb, wqb, Qb, nullptr);
  gemm_bf16_128x128<1><<<gemmBlocks, 256, 0, stream>>>(xb, wkb, Kb, nullptr);
  gemm_bf16_128x128<0><<<gemmBlocks, 256, 0, stream>>>(xb, wvb, Vt, nullptr);

  attn_kernel<<<(B_*H_*QT_)/8, 256, 0, stream>>>(Qb, Kb, Vt, ctx);

  gemm_bf16_128x128<2><<<gemmBlocks, 256, 0, stream>>>(ctx, wob, nullptr, out);
}
